// mnist_graph_pred_GNN_50122268344999
// MI455X (gfx1250) — matrix operand order repaired and checked by reading, not compiled, not run on hardware
//
#include <hip/hip_runtime.h>

// ---------------------------------------------------------------------------
// Types for CDNA5 WMMA
// ---------------------------------------------------------------------------
typedef __attribute__((ext_vector_type(16))) __bf16 v16bf;
typedef __attribute__((ext_vector_type(8)))  __bf16 v8bf;
typedef __attribute__((ext_vector_type(8)))  float  v8f;

#define BM 128
#define BN 128
#define BK 64

static __device__ __forceinline__ int imin(int a, int b) { return a < b ? a : b; }

// ---------------------------------------------------------------------------
// Utility kernels
// ---------------------------------------------------------------------------
__global__ void zero_f32_kernel(float* __restrict__ p, long long n) {
  long long i = (long long)blockIdx.x * blockDim.x + threadIdx.x;
  if (i < n) p[i] = 0.0f;
}

__global__ void f32_to_bf16_kernel(const float* __restrict__ in,
                                   __bf16* __restrict__ out, long long n) {
  long long i = (long long)blockIdx.x * blockDim.x + threadIdx.x;
  if (i < n) out[i] = (__bf16)in[i];
}

__global__ void deg_count_kernel(const long long* __restrict__ dst,
                                 float* __restrict__ deg, int E) {
  int e = blockIdx.x * blockDim.x + threadIdx.x;
  if (e < E) atomicAdd(&deg[(int)dst[e]], 1.0f);
}

// deg -> dinv_sqrt in place: dinv = rsqrt(deg + 1)
__global__ void rsqrt_inplace_kernel(float* __restrict__ deg, int n) {
  int i = blockIdx.x * blockDim.x + threadIdx.x;
  if (i < n) deg[i] = rsqrtf(deg[i] + 1.0f);
}

__global__ void edge_norm_kernel(const long long* __restrict__ src,
                                 const long long* __restrict__ dst,
                                 const float* __restrict__ dinv,
                                 float* __restrict__ norm, int E) {
  int e = blockIdx.x * blockDim.x + threadIdx.x;
  if (e < E) norm[e] = dinv[(int)src[e]] * dinv[(int)dst[e]];
}

// ---------------------------------------------------------------------------
// Tiled bf16 WMMA GEMM: C[MxN](f32) = A[MxK](bf16) @ B[KxN](bf16)
// Block = 256 threads = 8 waves; block tile 128x128, K-step 64 (16 WMMA/step).
// Out-of-range rows/cols handled by index clamping: the duplicated values only
// feed C rows/cols that are never stored, so no zero-fill predicates needed.
// K must be a multiple of 64 (true: 128, 256).
// ---------------------------------------------------------------------------
__global__ __launch_bounds__(256) void gemm_wmma_bf16_kernel(
    const __bf16* __restrict__ A, const __bf16* __restrict__ B,
    float* __restrict__ C, int M, int N, int K) {
  __shared__ alignas(32) __bf16 As[BM][BK];   // 16 KB, row-major [m][k]
  __shared__ alignas(32) __bf16 Bs[BN][BK];   // 16 KB, transposed [n][k]

  const int tid  = threadIdx.x;
  const int lane = tid & 31;
  const int wave = tid >> 5;
  const int wm   = (wave >> 1) * 32;    // wave M offset in block tile
  const int wn   = (wave & 1) * 64;     // wave N offset in block tile
  const int bm   = blockIdx.x * BM;
  const int bn   = blockIdx.y * BN;

  v8f acc[2][4] = {};

  // A staging: thread -> (row 0..127, 32-wide k half), clamped row
  const int arow = tid >> 1;
  const int akk  = (tid & 1) * 32;
  const int agr  = imin(bm + arow, M - 1);

  // B staging: thread -> (k row 0..63, 32-wide n chunk)
  const int bk  = tid >> 2;
  const int bn0 = (tid & 3) * 32;

  // Fragment addressing (A/B in contiguous k order: lane half h holds k = 16h..16h+15)
  const int lr    = lane & 15;
  const int abase = (lane >> 4) * 8;    // unused; A loads use kh below
  const int kh    = (lane >> 4) * 16;   // B: lanes>=16 take K-half +16

  for (int k0 = 0; k0 < K; k0 += BK) {
    // ---- Stage A tile: pure vector copy (b128 in, b128 out) ----
    {
      const __bf16* ap = A + (size_t)agr * K + (k0 + akk);
#pragma unroll
      for (int j = 0; j < 4; ++j)
        *(v8bf*)&As[arow][akk + 8 * j] = *(const v8bf*)(ap + 8 * j);
    }
    // ---- Stage B tile transposed: vector loads, scalar b16 scatter ----
    {
      if (bn + bn0 + 32 <= N) {
        const __bf16* bp = B + (size_t)(k0 + bk) * N + (bn + bn0);
#pragma unroll
        for (int j = 0; j < 4; ++j) {
          v8bf t = *(const v8bf*)(bp + 8 * j);
#pragma unroll
          for (int u = 0; u < 8; ++u) Bs[bn0 + 8 * j + u][bk] = t[u];
        }
      } else {
        const __bf16* brow = B + (size_t)(k0 + bk) * N;
#pragma unroll
        for (int j = 0; j < 32; ++j)
          Bs[bn0 + j][bk] = brow[imin(bn + bn0 + j, N - 1)];
      }
    }
    __syncthreads();

#pragma unroll
    for (int ks = 0; ks < BK; ks += 32) {
      // B fragments: contiguous 16 bf16 (32B) at [n][ks+kh]
      v16bf bf[4];
#pragma unroll
      for (int nt = 0; nt < 4; ++nt)
        bf[nt] = *(const v16bf*)&Bs[wn + nt * 16 + lr][ks + kh];

      // A fragments + WMMA
#pragma unroll
      for (int mt = 0; mt < 2; ++mt) {
        v8bf lo = *(const v8bf*)&As[wm + mt * 16 + lr][ks + kh];
        v8bf hi = *(const v8bf*)&As[wm + mt * 16 + lr][ks + kh + 8];
        v16bf af = __builtin_shufflevector(lo, hi, 0, 1, 2, 3, 4, 5, 6, 7,
                                           8, 9, 10, 11, 12, 13, 14, 15);
#pragma unroll
        for (int nt = 0; nt < 4; ++nt)
          acc[mt][nt] = __builtin_amdgcn_wmma_f32_16x16x32_bf16(
              false, af, false, bf[nt], (short)0, acc[mt][nt], false, false);
      }
    }
    __syncthreads();
  }

  // ---- Store C (ISA layout: VGPR r -> M = r + (lane>=16)*8, N = lane%16) ----
#pragma unroll
  for (int mt = 0; mt < 2; ++mt)
#pragma unroll
    for (int nt = 0; nt < 4; ++nt) {
      const int n = bn + wn + nt * 16 + (lane & 15);
#pragma unroll
      for (int r = 0; r < 8; ++r) {
        const int m = bm + wm + mt * 16 + (lane >> 4) * 8 + r;
        if (m < M && n < N) C[(size_t)m * N + n] = acc[mt][nt][r];
      }
    }
}

// ---------------------------------------------------------------------------
// Edge scatter: agg[dst] += H[src] * norm[e]   (float4 per thread)
// ---------------------------------------------------------------------------
__global__ void edge_scatter_kernel(const long long* __restrict__ src,
                                    const long long* __restrict__ dst,
                                    const float* __restrict__ norm,
                                    const float* __restrict__ H,
                                    float* __restrict__ agg, int E, int F) {
  const int vec = F >> 2;
  int t = blockIdx.x * blockDim.x + threadIdx.x;
  int e = t / vec;
  if (e >= E) return;
  int f4 = (t - e * vec) * 4;
  const int s = (int)src[e], d = (int)dst[e];
  const float w = norm[e];
  const float4 h = *(const float4*)(H + (size_t)s * F + f4);
  float* ap = agg + (size_t)d * F + f4;
  atomicAdd(ap + 0, h.x * w);
  atomicAdd(ap + 1, h.y * w);
  atomicAdd(ap + 2, h.z * w);
  atomicAdd(ap + 3, h.w * w);
}

// relu(agg + Hlin*dinv^2 + bias) -> bf16 (feeds next WMMA GEMM)
__global__ void fuse_relu_bf16_kernel(const float* __restrict__ agg,
                                      const float* __restrict__ Hlin,
                                      const float* __restrict__ dinv,
                                      const float* __restrict__ bias,
                                      __bf16* __restrict__ outbf,
                                      int Nn, int F) {
  long long idx = (long long)blockIdx.x * blockDim.x + threadIdx.x;
  if (idx >= (long long)Nn * F) return;
  int i = (int)(idx / F);
  int f = (int)(idx - (long long)i * F);
  float sn = dinv[i] * dinv[i];
  float v = agg[idx] + Hlin[idx] * sn + bias[f];
  outbf[idx] = (__bf16)fmaxf(v, 0.0f);
}

// relu(agg + Hlin*dinv^2 + bias) -> f32 in place (feeds classifier)
__global__ void fuse_relu_f32_kernel(float* __restrict__ agg,
                                     const float* __restrict__ Hlin,
                                     const float* __restrict__ dinv,
                                     const float* __restrict__ bias,
                                     int Nn, int F) {
  long long idx = (long long)blockIdx.x * blockDim.x + threadIdx.x;
  if (idx >= (long long)Nn * F) return;
  int i = (int)(idx / F);
  int f = (int)(idx - (long long)i * F);
  float sn = dinv[i] * dinv[i];
  float v = agg[idx] + Hlin[idx] * sn + bias[f];
  agg[idx] = fmaxf(v, 0.0f);
}

// ---------------------------------------------------------------------------
// Classifier + log_softmax: one wave32 per node, 10 classes, K=784
// ---------------------------------------------------------------------------
__global__ __launch_bounds__(256) void classifier_kernel(
    const float* __restrict__ H2, const float* __restrict__ Wc,
    const float* __restrict__ bc, float* __restrict__ out, int Nn) {
  const int lane = threadIdx.x & 31;
  const int wave = threadIdx.x >> 5;
  const int node = blockIdx.x * 8 + wave;
  if (node >= Nn) return;

  float acc[10];
#pragma unroll
  for (int c = 0; c < 10; ++c) acc[c] = 0.0f;

  const float* h = H2 + (size_t)node * 784;
  for (int j = lane; j < 784; j += 32) {
    const float hv = h[j];
    const float* wrow = Wc + j * 10;
#pragma unroll
    for (int c = 0; c < 10; ++c) acc[c] += hv * wrow[c];
  }
#pragma unroll
  for (int c = 0; c < 10; ++c) {
#pragma unroll
    for (int off = 16; off > 0; off >>= 1)
      acc[c] += __shfl_xor(acc[c], off, 32);
  }
  if (lane == 0) {
    float l[10], m = -INFINITY;
#pragma unroll
    for (int c = 0; c < 10; ++c) { l[c] = acc[c] + bc[c]; m = fmaxf(m, l[c]); }
    float s = 0.0f;
#pragma unroll
    for (int c = 0; c < 10; ++c) s += expf(l[c] - m);
    const float lse = m + logf(s);
    float* op = out + (size_t)node * 10;
#pragma unroll
    for (int c = 0; c < 10; ++c) op[c] = l[c] - lse;
  }
}

// ---------------------------------------------------------------------------
// Launch
// ---------------------------------------------------------------------------
extern "C" void kernel_launch(void* const* d_in, const int* in_sizes, int n_in,
                              void* d_out, int out_size, void* d_ws, size_t ws_size,
                              hipStream_t stream) {
  (void)in_sizes; (void)n_in; (void)out_size; (void)ws_size;

  const float*     x    = (const float*)d_in[0];
  const long long* eidx = (const long long*)d_in[1];
  const float*     W1   = (const float*)d_in[2];
  const float*     b1   = (const float*)d_in[3];
  const float*     W2   = (const float*)d_in[4];
  const float*     b2   = (const float*)d_in[5];
  const float*     Wc   = (const float*)d_in[6];
  const float*     bc   = (const float*)d_in[7];
  float*           out  = (float*)d_out;

  const int Nn = 50000, E = 400000;
  const int F1 = 128, F2 = 256, F3 = 784;
  const long long* src = eidx;
  const long long* dst = eidx + E;

  // Workspace carve-out
  char* wsp = (char*)d_ws;
  auto carve = [&](size_t bytes) -> char* {
    char* p = wsp;
    wsp += (bytes + 255) & ~(size_t)255;
    return p;
  };
  float*  deg   = (float*)carve((size_t)Nn * 4);            // -> dinv in place
  float*  norm  = (float*)carve((size_t)E * 4);
  __bf16* Xbf   = (__bf16*)carve((size_t)Nn * F1 * 2);
  __bf16* W1bf  = (__bf16*)carve((size_t)F1 * F2 * 2);
  __bf16* W2bf  = (__bf16*)carve((size_t)F2 * F3 * 2);
  __bf16* H1bf  = (__bf16*)carve((size_t)Nn * F2 * 2);
  float*  H1lin = (float*)carve((size_t)Nn * F2 * 4);
  float*  agg1  = (float*)carve((size_t)Nn * F2 * 4);
  float*  H2lin = (float*)carve((size_t)Nn * F3 * 4);
  float*  agg2  = (float*)carve((size_t)Nn * F3 * 4);

  const int TB = 256;
  auto cdiv = [](long long a, long long b) { return (int)((a + b - 1) / b); };

  // ---- Precision prep: bf16 copies of X, W1, W2 ----
  f32_to_bf16_kernel<<<cdiv((long long)Nn * F1, TB), TB, 0, stream>>>(x, Xbf, (long long)Nn * F1);
  f32_to_bf16_kernel<<<cdiv((long long)F1 * F2, TB), TB, 0, stream>>>(W1, W1bf, (long long)F1 * F2);
  f32_to_bf16_kernel<<<cdiv((long long)F2 * F3, TB), TB, 0, stream>>>(W2, W2bf, (long long)F2 * F3);

  // ---- Degrees + edge norms ----
  zero_f32_kernel<<<cdiv(Nn, TB), TB, 0, stream>>>(deg, Nn);
  deg_count_kernel<<<cdiv(E, TB), TB, 0, stream>>>(dst, deg, E);
  rsqrt_inplace_kernel<<<cdiv(Nn, TB), TB, 0, stream>>>(deg, Nn);
  edge_norm_kernel<<<cdiv(E, TB), TB, 0, stream>>>(src, dst, deg, norm, E);

  // ---- Layer 1: H1lin = X @ W1 ; aggregate ; relu -> bf16 ----
  {
    dim3 grid(cdiv(Nn, BM), cdiv(F2, BN));
    gemm_wmma_bf16_kernel<<<grid, TB, 0, stream>>>(Xbf, W1bf, H1lin, Nn, F2, F1);
  }
  zero_f32_kernel<<<cdiv((long long)Nn * F2, TB), TB, 0, stream>>>(agg1, (long long)Nn * F2);
  edge_scatter_kernel<<<cdiv((long long)E * (F2 / 4), TB), TB, 0, stream>>>(
      src, dst, norm, H1lin, agg1, E, F2);
  fuse_relu_bf16_kernel<<<cdiv((long long)Nn * F2, TB), TB, 0, stream>>>(
      agg1, H1lin, deg, b1, H1bf, Nn, F2);

  // ---- Layer 2: H2lin = H1 @ W2 ; aggregate ; relu (f32, in place) ----
  {
    dim3 grid(cdiv(Nn, BM), cdiv(F3, BN));
    gemm_wmma_bf16_kernel<<<grid, TB, 0, stream>>>(H1bf, W2bf, H2lin, Nn, F3, F2);
  }
  zero_f32_kernel<<<cdiv((long long)Nn * F3, TB), TB, 0, stream>>>(agg2, (long long)Nn * F3);
  edge_scatter_kernel<<<cdiv((long long)E * (F3 / 4), TB), TB, 0, stream>>>(
      src, dst, norm, H2lin, agg2, E, F3);
  fuse_relu_f32_kernel<<<cdiv((long long)Nn * F3, TB), TB, 0, stream>>>(
      agg2, H2lin, deg, b2, Nn, F3);

  // ---- Classifier + log_softmax ----
  classifier_kernel<<<cdiv(Nn, 8), TB, 0, stream>>>(agg2, Wc, bc, out, Nn);
}
